// GNNEncoder_12249246728338
// MI455X (gfx1250) — compile-verified, matrix operand order checked by reading, not run on hardware
//
#include <hip/hip_runtime.h>
#include <hip/hip_bf16.h>
#include <math.h>

typedef __attribute__((ext_vector_type(16))) __bf16 v16bf;
typedef __attribute__((ext_vector_type(8)))  float  v8f;

#define NN      4096
#define EE      131072
#define ETOT    (EE + NN)
#define IN_DIM  128
#define HID     64
#define H1N     4
#define DM      256
#define NHEAD   4
#define HD      64
#define DFF     2048
#define OUTD    128

// -------- CDNA5 feature detection (compile-clean fallbacks) --------
#if defined(__has_builtin)
#  if __has_builtin(__builtin_amdgcn_tensor_load_to_lds) && \
      __has_builtin(__builtin_amdgcn_s_wait_tensorcnt)
#    define HAVE_TDM 1
#  endif
#  if __has_builtin(__builtin_amdgcn_global_load_async_to_lds_b128) && \
      __has_builtin(__builtin_amdgcn_s_wait_asynccnt)
#    define HAVE_ASYNC 1
#  endif
#endif
#ifndef HAVE_TDM
#define HAVE_TDM 0
#endif
#ifndef HAVE_ASYNC
#define HAVE_ASYNC 0
#endif

typedef __attribute__((ext_vector_type(4))) int      v4i_t;
typedef __attribute__((ext_vector_type(8))) int      v8i_t;
typedef __attribute__((ext_vector_type(4))) unsigned v4u_t;

// ---------------------------------------------------------------- helpers

__device__ __forceinline__ v8f wmma_bf16(v16bf a, v16bf b, v8f c) {
  return __builtin_amdgcn_wmma_f32_16x16x32_bf16(
      false, a, false, b, (short)0, c, false, false);
}

// Fragment where K runs contiguously along an LDS row.
// 16-bit A/B 16x32 layout: lane m (0-15) holds row m; hi = lane>>4;
// pair j -> K0 = (j>>2)*16 + hi*8 + (j&3)*2  => two contiguous 16B chunks.
__device__ __forceinline__ v16bf frag_row(const __bf16* rowp, int hi) {
  v16bf f;
#pragma unroll
  for (int j = 0; j < 8; ++j) {
    int k0 = ((j >> 2) << 4) | (hi << 3) | ((j & 3) << 1);
    f[2 * j]     = rowp[k0];
    f[2 * j + 1] = rowp[k0 + 1];
  }
  return f;
}

// 16-byte global->LDS stage: async-to-LDS if available, else direct copy.
__device__ __forceinline__ void stage16(const __bf16* g, __bf16* l) {
#if HAVE_ASYNC
  __builtin_amdgcn_global_load_async_to_lds_b128(
      (__attribute__((address_space(1))) v4i_t*)(unsigned long long)(size_t)g,
      (__attribute__((address_space(3))) v4i_t*)(unsigned)(size_t)l, 0, 0);
#else
  *(uint4*)l = *(const uint4*)g;
#endif
}
__device__ __forceinline__ void stage_fence() {
#if HAVE_ASYNC
  __builtin_amdgcn_s_wait_asynccnt(0);
#endif
}

#if HAVE_TDM
// Tensor DMA descriptor for a 64-row x 32-col bf16 tile (global row stride =
// K elems) landing in LDS with 64B data + 16B pad rows (stride 40 bf16).
// Invariant words built once; per-issue we patch only the addresses.
struct TdmDesc {
  v4u_t g0;
  v8i_t g1;
};
__device__ __forceinline__ TdmDesc tdm_make(int K) {
  TdmDesc d;
  d.g0[0] = 1u;                                            // count=1
  d.g0[1] = 0u; d.g0[2] = 0u; d.g0[3] = 0u;                // patched per issue
  d.g1[0] = (1 << 16) | (1 << 20) | (3 << 22) | (3 << 25); // dsize=2B,pad_en,
                                                           // interval=16DW,amt=4DW
  d.g1[1] = (int)0xffff0000u;      // tensor_dim0 lo16 (dim0 = 0x7fffffff)
  d.g1[2] = (int)0xffff7fffu;      // dim1 lo16 | dim0 hi16
  d.g1[3] = (32 << 16) | 0x7fff;   // tile_dim0=32 | dim1 hi16
  d.g1[4] = 64;                    // tile_dim1=64 (tile_dim2=0)
  d.g1[5] = K;                     // tensor_dim0_stride lo32 (elements)
  d.g1[6] = 0;
  d.g1[7] = 0;
  return d;
}
__device__ __forceinline__ void tdm_issue(TdmDesc d, const __bf16* gsrc, void* ldst) {
  unsigned long long ga = (unsigned long long)(size_t)gsrc;
  d.g0[1] = (unsigned)(size_t)ldst;            // LDS byte offset
  d.g0[2] = (unsigned)ga;                      // global_addr lo
  d.g0[3] = (unsigned)(ga >> 32) | (2u << 30); // global hi | type=2
  v4i_t z4 = {0, 0, 0, 0};
  v8i_t z8 = {0, 0, 0, 0, 0, 0, 0, 0};
  __builtin_amdgcn_tensor_load_to_lds(d.g0, d.g1, z4, z4, z8, 0);
}
#endif

// monotone float <-> uint mapping for atomicMax on floats
__device__ __forceinline__ unsigned encF(float f) {
  unsigned u = __float_as_uint(f);
  return (u & 0x80000000u) ? ~u : (u | 0x80000000u);
}
__device__ __forceinline__ float decF(unsigned u) {
  unsigned v = (u & 0x80000000u) ? (u & 0x7fffffffu) : ~u;
  return __uint_as_float(v);
}

// ---------------------------------------------------------------- GEMM
// C[M,N] = A[M,K] @ W[N,K]^T + bias; bf16 operands, f32 WMMA accum.
// block = 128 threads (4 waves); tile 64x64; K step 32; TDM double-buffered.
template <int ACT>  // 0 = none, 1 = relu
__global__ void k_gemm(const __bf16* __restrict__ A, const __bf16* __restrict__ W,
                       const float* __restrict__ bias, float* __restrict__ C,
                       __bf16* __restrict__ Cb, int M, int Nn, int K) {
  __shared__ __bf16 As[2][64][40];
  __shared__ __bf16 Ws[2][64][40];
  const int tid  = threadIdx.x;
  const int wave = tid >> 5, lane = tid & 31;
  const int ml   = lane & 15, hi = lane >> 4;
  const int row0 = blockIdx.y * 64, col0 = blockIdx.x * 64;

  v8f acc[4] = {v8f{}, v8f{}, v8f{}, v8f{}};
  const int nsteps = K >> 5;
  const __bf16* Abase = A + (size_t)row0 * K;
  const __bf16* Wbase = W + (size_t)col0 * K;

#if HAVE_TDM
  TdmDesc dsc = tdm_make(K);
  if (wave == 0) {
    tdm_issue(dsc, Abase, &As[0][0][0]);
    tdm_issue(dsc, Wbase, &Ws[0][0][0]);
  }
  for (int s = 0; s < nsteps; ++s) {
    const int buf = s & 1;
    if (wave == 0) {
      if (s + 1 < nsteps) {
        tdm_issue(dsc, Abase + (s + 1) * 32, &As[buf ^ 1][0][0]);
        tdm_issue(dsc, Wbase + (s + 1) * 32, &Ws[buf ^ 1][0][0]);
        __builtin_amdgcn_s_wait_tensorcnt(2);  // current tiles landed
      } else {
        __builtin_amdgcn_s_wait_tensorcnt(0);
      }
    }
    __syncthreads();
    v16bf a = frag_row(&As[buf][wave * 16 + ml][0], hi);
#pragma unroll
    for (int t = 0; t < 4; ++t) {
      v16bf b = frag_row(&Ws[buf][t * 16 + ml][0], hi);
      acc[t] = wmma_bf16(a, b, acc[t]);
    }
    __syncthreads();
  }
#else
  const int r = tid >> 1, p16 = (tid & 1) * 16;  // 64 rows x 2 halves = 128 thr
  for (int s = 0; s < nsteps; ++s) {
    const int kb = s * 32;
    if (r < 64) {
      stage16(&Abase[(size_t)r * K + kb + p16], &As[0][r][p16]);
      stage16(&Wbase[(size_t)r * K + kb + p16], &Ws[0][r][p16]);
    }
    stage_fence();
    __syncthreads();
    v16bf a = frag_row(&As[0][wave * 16 + ml][0], hi);
#pragma unroll
    for (int t = 0; t < 4; ++t) {
      v16bf b = frag_row(&Ws[0][t * 16 + ml][0], hi);
      acc[t] = wmma_bf16(a, b, acc[t]);
    }
    __syncthreads();
  }
#endif

#pragma unroll
  for (int t = 0; t < 4; ++t) {
    int col = col0 + t * 16 + ml;
    float bv = bias ? bias[col] : 0.0f;
#pragma unroll
    for (int r8 = 0; r8 < 8; ++r8) {
      int row = row0 + wave * 16 + hi * 8 + r8;
      float v = acc[t][r8] + bv;
      if (ACT == 1) v = fmaxf(v, 0.0f);
      if (C)  C[(size_t)row * Nn + col]  = v;
      if (Cb) Cb[(size_t)row * Nn + col] = (__bf16)v;
    }
  }
}

// ---------------------------------------------------------------- flash attention
// grid (N/16, NHEAD), block = 32 (one wave). bf16 QKV in, bf16 O out.
__global__ void k_attn(const __bf16* __restrict__ qkvb, __bf16* __restrict__ attb) {
  __shared__ __bf16 Qs[16][72];
  __shared__ __bf16 Ks[32][72];
  __shared__ __bf16 Vt[64][40];   // transposed: Vt[d][k]
  __shared__ __bf16 Ps[16][40];
  __shared__ float  ms[16], ls[16];

  const int h    = blockIdx.y;
  const int q0   = blockIdx.x * 16;
  const int lane = threadIdx.x;
  const int nl   = lane & 15, hi = lane >> 4;
  const int r0   = lane >> 3, p8 = (lane & 7) * 8;  // per-lane chunk geometry
  const float scale = 0.125f;  // 1/sqrt(64)
  const int QS = 3 * DM;       // qkv row stride (bf16 elems)

  // Q tile: 16 rows x 64 bf16 = 128 chunks of 16B (4 per lane)
  {
    const __bf16* gq = qkvb + (size_t)(q0 + r0) * QS + h * HD + p8;
#pragma unroll
    for (int i = 0; i < 4; ++i)
      stage16(gq + (size_t)i * 4 * QS, &Qs[i * 4 + r0][p8]);
  }
  stage_fence();
  if (lane < 16) { ms[lane] = -1e30f; ls[lane] = 0.0f; }
  __syncthreads();

  v16bf qa0 = frag_row(&Qs[nl][0], hi);
  v16bf qa1 = frag_row(&Qs[nl][32], hi);
  v8f o0{}, o1{}, o2{}, o3{};

  // loop-invariant per-lane staging pointers, advanced by 32 rows per tile
  const __bf16* gk  = qkvb + (size_t)r0 * QS + DM + h * HD + p8;
  const __bf16* gv0 = qkvb + 2 * DM + h * HD + lane;
  const __bf16* gv1 = gv0 + 32;

  for (int kt = 0; kt < NN; kt += 32) {
    // K tile: 32 rows x 64 bf16 = 256 chunks of 16B (async gather to LDS)
#pragma unroll
    for (int i = 0; i < 8; ++i)
      stage16(gk + (size_t)i * 4 * QS, &Ks[i * 4 + r0][p8]);
    gk += (size_t)32 * QS;
    // V tile, stored transposed: lane owns dims (lane, lane+32)
#pragma unroll
    for (int k = 0; k < 32; ++k) {
      Vt[lane][k]      = gv0[(size_t)k * QS];
      Vt[lane + 32][k] = gv1[(size_t)k * QS];
    }
    gv0 += (size_t)32 * QS;
    gv1 += (size_t)32 * QS;
    stage_fence();
    __syncthreads();

    // S = Q @ K^T : 16x32 as two 16x16 tiles, K-dim 64 split in two WMMAs
    v8f s0{}, s1{};
    s0 = wmma_bf16(qa0, frag_row(&Ks[nl][0], hi), s0);
    s0 = wmma_bf16(qa1, frag_row(&Ks[nl][32], hi), s0);
    s1 = wmma_bf16(qa0, frag_row(&Ks[16 + nl][0], hi), s1);
    s1 = wmma_bf16(qa1, frag_row(&Ks[16 + nl][32], hi), s1);

    // online softmax; row = hi*8 + r; columns across 16 lanes
#pragma unroll
    for (int r = 0; r < 8; ++r) {
      int row = hi * 8 + r;
      float v0 = s0[r] * scale, v1 = s1[r] * scale;
      float mx = fmaxf(v0, v1);
#pragma unroll
      for (int off = 8; off; off >>= 1) mx = fmaxf(mx, __shfl_xor(mx, off, 16));
      float mold = ms[row];
      float mnew = fmaxf(mold, mx);
      float p0 = __expf(v0 - mnew), p1 = __expf(v1 - mnew);
      float ps = p0 + p1;
#pragma unroll
      for (int off = 8; off; off >>= 1) ps += __shfl_xor(ps, off, 16);
      float cf = __expf(mold - mnew);
      if (nl == 0) { ms[row] = mnew; ls[row] = ls[row] * cf + ps; }
      Ps[row][nl]      = (__bf16)p0;
      Ps[row][16 + nl] = (__bf16)p1;
      o0[r] *= cf; o1[r] *= cf; o2[r] *= cf; o3[r] *= cf;
    }
    __syncthreads();

    // O += P(16x32) @ V(32x64); V fragments are contiguous rows of Vt
    v16bf pa = frag_row(&Ps[nl][0], hi);
    o0 = wmma_bf16(pa, frag_row(&Vt[0  + nl][0], hi), o0);
    o1 = wmma_bf16(pa, frag_row(&Vt[16 + nl][0], hi), o1);
    o2 = wmma_bf16(pa, frag_row(&Vt[32 + nl][0], hi), o2);
    o3 = wmma_bf16(pa, frag_row(&Vt[48 + nl][0], hi), o3);
    __syncthreads();
  }

#pragma unroll
  for (int r = 0; r < 8; ++r) {
    int row = q0 + hi * 8 + r;
    float inv = 1.0f / (ls[hi * 8 + r] + 1e-16f);
    attb[(size_t)row * DM + h * HD + 0  + nl] = (__bf16)(o0[r] * inv);
    attb[(size_t)row * DM + h * HD + 16 + nl] = (__bf16)(o1[r] * inv);
    attb[(size_t)row * DM + h * HD + 32 + nl] = (__bf16)(o2[r] * inv);
    attb[(size_t)row * DM + h * HD + 48 + nl] = (__bf16)(o3[r] * inv);
  }
}

// ---------------------------------------------------------------- LayerNorm(x + r)
__global__ void k_ln(const float* __restrict__ x, const float* __restrict__ r,
                     const float* __restrict__ g, const float* __restrict__ b,
                     float* __restrict__ out, __bf16* __restrict__ outb, int n) {
  int wave = threadIdx.x >> 5, lane = threadIdx.x & 31;
  int row = blockIdx.x * 4 + wave;
  if (row >= n) return;
  const float* xr = x + (size_t)row * DM;
  const float* rr = r + (size_t)row * DM;
  float v[8], s = 0.0f;
#pragma unroll
  for (int i = 0; i < 8; ++i) { v[i] = xr[i * 32 + lane] + rr[i * 32 + lane]; s += v[i]; }
#pragma unroll
  for (int off = 16; off; off >>= 1) s += __shfl_xor(s, off, 32);
  float mean = s * (1.0f / DM);
  float var = 0.0f;
#pragma unroll
  for (int i = 0; i < 8; ++i) { float d = v[i] - mean; var += d * d; }
#pragma unroll
  for (int off = 16; off; off >>= 1) var += __shfl_xor(var, off, 32);
  float rstd = rsqrtf(var * (1.0f / DM) + 1e-5f);
#pragma unroll
  for (int i = 0; i < 8; ++i) {
    int c = i * 32 + lane;
    float y = (v[i] - mean) * rstd * g[c] + b[c];
    if (out)  out[(size_t)row * DM + c]  = y;
    if (outb) outb[(size_t)row * DM + c] = (__bf16)y;
  }
}

// ---------------------------------------------------------------- GAT pieces
template <int H, int C>
__global__ void k_alphas(const float* __restrict__ P, const float* __restrict__ asrc,
                         const float* __restrict__ adst, float* __restrict__ als,
                         float* __restrict__ ald) {
  int i = blockIdx.x * blockDim.x + threadIdx.x;
  if (i >= NN * H) return;
  int n = i / H, h = i % H;
  float s = 0.0f, d = 0.0f;
  const float* p = P + (size_t)n * (H * C) + h * C;
#pragma unroll 4
  for (int c = 0; c < C; ++c) { s += p[c] * asrc[h * C + c]; d += p[c] * adst[h * C + c]; }
  als[i] = s; ald[i] = d;
}

template <int H>
__global__ void k_edge_max(const int* __restrict__ ei, const float* __restrict__ als,
                           const float* __restrict__ ald, float* __restrict__ elog,
                           unsigned* __restrict__ segm) {
  int e = blockIdx.x * blockDim.x + threadIdx.x;
  if (e >= ETOT) return;
  int src, dst;
  if (e < EE) { src = ei[e]; dst = ei[EE + e]; } else { src = dst = e - EE; }
#pragma unroll
  for (int h = 0; h < H; ++h) {
    float l = als[src * H + h] + ald[dst * H + h];
    l = (l > 0.0f) ? l : 0.2f * l;  // leaky_relu(0.2)
    elog[(size_t)e * H + h] = l;
    atomicMax(&segm[dst * H + h], encF(l));
  }
}

template <int H>
__global__ void k_edge_exp(const int* __restrict__ ei, float* __restrict__ elog,
                           const unsigned* __restrict__ segm, float* __restrict__ segs) {
  int e = blockIdx.x * blockDim.x + threadIdx.x;
  if (e >= ETOT) return;
  int dst = (e < EE) ? ei[EE + e] : (e - EE);
#pragma unroll
  for (int h = 0; h < H; ++h) {
    float m  = decF(segm[dst * H + h]);
    float ex = __expf(elog[(size_t)e * H + h] - m);
    elog[(size_t)e * H + h] = ex;
    atomicAdd(&segs[dst * H + h], ex);
  }
}

template <int H, int C>
__global__ void k_edge_agg(const int* __restrict__ ei, const float* __restrict__ P,
                           const float* __restrict__ elog, const float* __restrict__ segs,
                           float* __restrict__ out) {
  int wave = threadIdx.x >> 5, lane = threadIdx.x & 31;
  int e = blockIdx.x * 4 + wave;
  if (e >= ETOT) return;
  int src, dst;
  if (e < EE) { src = ei[e]; dst = ei[EE + e]; } else { src = dst = e - EE; }
#pragma unroll
  for (int c = lane; c < H * C; c += 32) {
    int h = c / C;
    float alpha = elog[(size_t)e * H + h] / (segs[dst * H + h] + 1e-16f);
    atomicAdd(&out[(size_t)dst * (H * C) + c], P[(size_t)src * (H * C) + c] * alpha);
  }
}

// ---------------------------------------------------------------- elementwise
__global__ void k_cvt(const float* __restrict__ in, __bf16* __restrict__ out, int n) {
  int i = (blockIdx.x * blockDim.x + threadIdx.x) * 4;
  if (i + 3 >= n) { for (; i < n; ++i) out[i] = (__bf16)in[i]; return; }
  float4 v = *(const float4*)&in[i];
  out[i] = (__bf16)v.x; out[i + 1] = (__bf16)v.y;
  out[i + 2] = (__bf16)v.z; out[i + 3] = (__bf16)v.w;
}
__global__ void k_fill_f(float* p, float v, int n) {
  int i = blockIdx.x * blockDim.x + threadIdx.x;
  if (i < n) p[i] = v;
}
__global__ void k_fill_u(unsigned* p, unsigned v, int n) {
  int i = blockIdx.x * blockDim.x + threadIdx.x;
  if (i < n) p[i] = v;
}
__global__ void k_bias_elu(const float* __restrict__ g, const float* __restrict__ bias,
                           float* __restrict__ out, __bf16* __restrict__ outb,
                           int C, int n) {
  int i = blockIdx.x * blockDim.x + threadIdx.x;
  if (i >= n) return;
  float v = g[i] + bias[i & (C - 1)];
  v = (v > 0.0f) ? v : expm1f(v);
  out[i]  = v;
  outb[i] = (__bf16)v;
}
__global__ void k_bias_add(float* __restrict__ out, const float* __restrict__ bias,
                           int C, int n) {
  int i = blockIdx.x * blockDim.x + threadIdx.x;
  if (i < n) out[i] += bias[i & (C - 1)];
}

// ---------------------------------------------------------------- launch
extern "C" void kernel_launch(void* const* d_in, const int* in_sizes, int n_in,
                              void* d_out, int out_size, void* d_ws, size_t ws_size,
                              hipStream_t stream) {
  const float* x        = (const float*)d_in[0];
  const int*   ei       = (const int*)d_in[1];
  const float* gat1_w   = (const float*)d_in[2];
  const float* gat1_as  = (const float*)d_in[3];
  const float* gat1_ad  = (const float*)d_in[4];
  const float* gat1_b   = (const float*)d_in[5];
  const float* inproj_w = (const float*)d_in[6];
  const float* inproj_b = (const float*)d_in[7];
  const float* oproj_w  = (const float*)d_in[8];
  const float* oproj_b  = (const float*)d_in[9];
  const float* lin1_w   = (const float*)d_in[10];
  const float* lin1_b   = (const float*)d_in[11];
  const float* lin2_w   = (const float*)d_in[12];
  const float* lin2_b   = (const float*)d_in[13];
  const float* ln1_g    = (const float*)d_in[14];
  const float* ln1_b    = (const float*)d_in[15];
  const float* ln2_g    = (const float*)d_in[16];
  const float* ln2_b    = (const float*)d_in[17];
  const float* gat2_w   = (const float*)d_in[18];
  const float* gat2_as  = (const float*)d_in[19];
  const float* gat2_ad  = (const float*)d_in[20];
  const float* gat2_b   = (const float*)d_in[21];
  float* out = (float*)d_out;

  // ---- fp32 workspace
  float* ws = (float*)d_ws;
  size_t o = 0;
  float* P1  = ws + o; o += (size_t)NN * DM;
  float* G1  = ws + o; o += (size_t)NN * DM;
  float* Hbf = ws + o; o += (size_t)NN * DM;
  float* AO  = ws + o; o += (size_t)NN * DM;
  float* HL1 = ws + o; o += (size_t)NN * DM;
  float* FF2 = ws + o; o += (size_t)NN * DM;
  float* P2  = ws + o; o += (size_t)NN * OUTD;
  float* ALS = ws + o; o += (size_t)NN * H1N;
  float* ALD = ws + o; o += (size_t)NN * H1N;
  unsigned* SEGM = (unsigned*)(ws + o); o += (size_t)NN * H1N;
  float* SEGS = ws + o; o += (size_t)NN * H1N;
  float* ELOG = ws + o; o += (size_t)ETOT * H1N;
  // ---- bf16 workspace (16B aligned: all prior sizes are multiples of 8 floats)
  __bf16* bw = (__bf16*)(ws + o);
  size_t ob = 0;
  __bf16* XB   = bw + ob; ob += (size_t)NN * IN_DIM;
  __bf16* W1B  = bw + ob; ob += (size_t)DM * IN_DIM;
  __bf16* WQB  = bw + ob; ob += (size_t)3 * DM * DM;
  __bf16* WOB  = bw + ob; ob += (size_t)DM * DM;
  __bf16* L1B  = bw + ob; ob += (size_t)DFF * DM;
  __bf16* L2B  = bw + ob; ob += (size_t)DM * DFF;
  __bf16* W2B  = bw + ob; ob += (size_t)OUTD * DM;
  __bf16* HBb  = bw + ob; ob += (size_t)NN * DM;
  __bf16* QKVB = bw + ob; ob += (size_t)NN * 3 * DM;
  __bf16* ATTb = bw + ob; ob += (size_t)NN * DM;
  __bf16* HL1b = bw + ob; ob += (size_t)NN * DM;
  __bf16* FF1b = bw + ob; ob += (size_t)NN * DFF;
  __bf16* HL2b = bw + ob; ob += (size_t)NN * DM;

  const unsigned NEG_INF_KEY = 0x007FFFFFu;  // encF(-inf)
  const int TB = 256;
#define GRID1(n) dim3(((n) + TB - 1) / TB), dim3(TB), 0, stream

  // ---- one-time bf16 conversion of inputs/weights
  k_cvt<<<GRID1((NN * IN_DIM) / 4)>>>(x, XB, NN * IN_DIM);
  k_cvt<<<GRID1((DM * IN_DIM) / 4)>>>(gat1_w, W1B, DM * IN_DIM);
  k_cvt<<<GRID1((3 * DM * DM) / 4)>>>(inproj_w, WQB, 3 * DM * DM);
  k_cvt<<<GRID1((DM * DM) / 4)>>>(oproj_w, WOB, DM * DM);
  k_cvt<<<GRID1((DFF * DM) / 4)>>>(lin1_w, L1B, DFF * DM);
  k_cvt<<<GRID1((DM * DFF) / 4)>>>(lin2_w, L2B, DM * DFF);
  k_cvt<<<GRID1((OUTD * DM) / 4)>>>(gat2_w, W2B, OUTD * DM);

  // ---- GAT layer 1
  k_gemm<0><<<dim3(DM / 64, NN / 64), 128, 0, stream>>>(XB, W1B, nullptr, P1, nullptr, NN, DM, IN_DIM);
  k_alphas<H1N, HID><<<GRID1(NN * H1N)>>>(P1, gat1_as, gat1_ad, ALS, ALD);
  k_fill_f<<<GRID1(NN * DM)>>>(G1, 0.0f, NN * DM);
  k_fill_u<<<GRID1(NN * H1N)>>>(SEGM, NEG_INF_KEY, NN * H1N);
  k_fill_f<<<GRID1(NN * H1N)>>>(SEGS, 0.0f, NN * H1N);
  k_edge_max<H1N><<<GRID1(ETOT)>>>(ei, ALS, ALD, ELOG, SEGM);
  k_edge_exp<H1N><<<GRID1(ETOT)>>>(ei, ELOG, SEGM, SEGS);
  k_edge_agg<H1N, HID><<<dim3((ETOT + 3) / 4), 128, 0, stream>>>(ei, P1, ELOG, SEGS, G1);
  k_bias_elu<<<GRID1(NN * DM)>>>(G1, gat1_b, Hbf, HBb, DM, NN * DM);

  // ---- Transformer encoder layer
  k_gemm<0><<<dim3(3 * DM / 64, NN / 64), 128, 0, stream>>>(HBb, WQB, inproj_b, nullptr, QKVB, NN, 3 * DM, DM);
  k_attn<<<dim3(NN / 16, NHEAD), 32, 0, stream>>>(QKVB, ATTb);
  k_gemm<0><<<dim3(DM / 64, NN / 64), 128, 0, stream>>>(ATTb, WOB, oproj_b, AO, nullptr, NN, DM, DM);
  k_ln<<<dim3(NN / 4), 128, 0, stream>>>(Hbf, AO, ln1_g, ln1_b, HL1, HL1b, NN);
  k_gemm<1><<<dim3(DFF / 64, NN / 64), 128, 0, stream>>>(HL1b, L1B, lin1_b, nullptr, FF1b, NN, DFF, DM);
  k_gemm<0><<<dim3(DM / 64, NN / 64), 128, 0, stream>>>(FF1b, L2B, lin2_b, FF2, nullptr, NN, DM, DFF);
  k_ln<<<dim3(NN / 4), 128, 0, stream>>>(HL1, FF2, ln2_g, ln2_b, nullptr, HL2b, NN);

  // ---- GAT layer 2 (1 head, 128 ch) -> d_out
  k_gemm<0><<<dim3(OUTD / 64, NN / 64), 128, 0, stream>>>(HL2b, W2B, nullptr, P2, nullptr, NN, OUTD, DM);
  k_alphas<1, OUTD><<<GRID1(NN)>>>(P2, gat2_as, gat2_ad, ALS, ALD);
  k_fill_f<<<GRID1(NN * OUTD)>>>(out, 0.0f, NN * OUTD);
  k_fill_u<<<GRID1(NN)>>>(SEGM, NEG_INF_KEY, NN);
  k_fill_f<<<GRID1(NN)>>>(SEGS, 0.0f, NN);
  k_edge_max<1><<<GRID1(ETOT)>>>(ei, ALS, ALD, ELOG, SEGM);
  k_edge_exp<1><<<GRID1(ETOT)>>>(ei, ELOG, SEGM, SEGS);
  k_edge_agg<1, OUTD><<<dim3((ETOT + 3) / 4), 128, 0, stream>>>(ei, P2, ELOG, SEGS, out);
  k_bias_add<<<GRID1(NN * OUTD)>>>(out, gat2_b, OUTD, NN * OUTD);

  (void)in_sizes; (void)n_in; (void)out_size; (void)ws_size;
#undef GRID1
}
